// BitStackLinear_730144440922
// MI455X (gfx1250) — compile-verified, matrix operand order checked by reading, not run on hardware
//
#include <hip/hip_runtime.h>

#define IN_F   4096
#define OUT_F  11008
#define W_BIT  4
#define RK     16
#define M_TOT  8192   // 4 * 2048

typedef __attribute__((ext_vector_type(16))) _Float16 v16h;
typedef __attribute__((ext_vector_type(8)))  _Float16 v8h;
typedef __attribute__((ext_vector_type(4)))  _Float16 v4h;
typedef __attribute__((ext_vector_type(8)))  float    v8f;
typedef __attribute__((ext_vector_type(4)))  float    v4f;

// ---------------------------------------------------------------------------
// Kernel 1: convert x (f32) -> f16 workspace. Sizes are exact multiples of
// 1024 elements per block, so no bounds checks needed.
// ---------------------------------------------------------------------------
__global__ __launch_bounds__(256)
void cvt_x_kernel(const float* __restrict__ x, _Float16* __restrict__ x16) {
    size_t i = ((size_t)blockIdx.x * 256 + threadIdx.x) * 4;
    v4f v = *(const v4f*)(x + i);
    v4h o;
    o.x = (_Float16)v.x;
    o.y = (_Float16)v.y;
    o.z = (_Float16)v.z;
    o.w = (_Float16)v.w;
    *(v4h*)(x16 + i) = o;
}

// ---------------------------------------------------------------------------
// Kernel 2: reconstruct w[n,k] = sum_i sign_i(n,k) * (u_i[n,:] . vt_i[:,k])
// One block per row n. u-row (4 planes x 16) staged in LDS. vt is only 1 MB
// total so it stays L2-resident across the 11008 blocks.
// ---------------------------------------------------------------------------
__global__ __launch_bounds__(256)
void recon_w_kernel(const int* __restrict__ qw, const float* __restrict__ u,
                    const float* __restrict__ vt, _Float16* __restrict__ w16) {
    const int n = blockIdx.x;
    __shared__ float su[W_BIT * RK];
    if (threadIdx.x < W_BIT * RK) {
        int i = threadIdx.x >> 4;   // plane
        int j = threadIdx.x & 15;   // rank index
        su[threadIdx.x] = u[(size_t)i * OUT_F * RK + (size_t)n * RK + j];
    }
    __syncthreads();

    const size_t rowByteBase = (size_t)n * (IN_F / 8);   // 512 packed bytes/row
    for (int k = threadIdx.x; k < IN_F; k += 256) {
        float acc = 0.0f;
#pragma unroll
        for (int i = 0; i < W_BIT; ++i) {
            int byte = qw[(size_t)i * ((size_t)OUT_F * (IN_F / 8)) + rowByteBase + (k >> 3)];
            float s = ((byte >> (k & 7)) & 1) ? 1.0f : -1.0f;
            float dot = 0.0f;
#pragma unroll
            for (int j = 0; j < RK; ++j)
                dot += su[i * RK + j] * vt[(size_t)i * RK * IN_F + (size_t)j * IN_F + k];
            acc += s * dot;
        }
        w16[(size_t)n * IN_F + k] = (_Float16)acc;
    }
}

// ---------------------------------------------------------------------------
// Kernel 3: C[m,n] = sum_k A16[m,k] * B16[n,k]  (NT GEMM, f16 in / f32 out)
// Block = 256 threads = 8 waves arranged 2(M) x 4(N).
// Block tile: 128(M) x 256(N). Wave tile: 64(M) x 64(N) = 4x4 WMMA tiles.
// Per 32-deep K step: 8 fragment loads feed 16 v_wmma_f32_16x16x32_f16
// (0.5 loads/WMMA — 1.5x better fragment-bandwidth ratio than a 32x64 tile).
//
// 16-bit A fragment layout (ISA 7.12.2): lane = hs*16 + r holds row M=r,
//   halves 0..7  -> K = kb + hs*8 + {0..7}
//   halves 8..15 -> K = kb + 16 + hs*8 + {0..7}
// B (32x16, K-major columns) mirrors this with N = lane&15; since w is
// row-major [N,K], B fragment loads are identical in form to A loads.
// ---------------------------------------------------------------------------
__global__ __launch_bounds__(256, 1)
void gemm_f16_kernel(const _Float16* __restrict__ A,   // [M_TOT, IN_F]
                     const _Float16* __restrict__ B,   // [OUT_F, IN_F]
                     float* __restrict__ C) {          // [M_TOT, OUT_F]
    const int lane  = threadIdx.x & 31;
    const int wave  = threadIdx.x >> 5;
    const int waveM = wave >> 2;          // 0..1
    const int waveN = wave & 3;           // 0..3
    const int row   = lane & 15;
    const int hs    = lane >> 4;          // half-select (K sub-block)

    const int nBase = blockIdx.x * 256 + waveN * 64;
    const int mBase = blockIdx.y * 128 + waveM * 64;

    union Frag { v16h v; v8h h[2]; };

    v8f acc[4][4];
#pragma unroll
    for (int mi = 0; mi < 4; ++mi)
#pragma unroll
        for (int ni = 0; ni < 4; ++ni)
            acc[mi][ni] = (v8f){0.f, 0.f, 0.f, 0.f, 0.f, 0.f, 0.f, 0.f};

    const _Float16* aPtr = A + (size_t)(mBase + row) * IN_F + hs * 8;
    const _Float16* bPtr = B + (size_t)(nBase + row) * IN_F + hs * 8;

    for (int k = 0; k < IN_F; k += 32) {
        Frag a[4];
#pragma unroll
        for (int mi = 0; mi < 4; ++mi) {
            const _Float16* p = aPtr + (size_t)mi * 16 * IN_F + k;
            a[mi].h[0] = *(const v8h*)(p);
            a[mi].h[1] = *(const v8h*)(p + 16);
        }
#pragma unroll
        for (int ni = 0; ni < 4; ++ni) {
            Frag b;
            const _Float16* p = bPtr + (size_t)ni * 16 * IN_F + k;
            b.h[0] = *(const v8h*)(p);
            b.h[1] = *(const v8h*)(p + 16);
#pragma unroll
            for (int mi = 0; mi < 4; ++mi)
                acc[mi][ni] = __builtin_amdgcn_wmma_f32_16x16x32_f16(
                    /*neg_a=*/false, a[mi].v,
                    /*neg_b=*/false, b.v,
                    /*c_mod=*/(short)0, acc[mi][ni],
                    /*reuse_a=*/false, /*reuse_b=*/false);
        }
    }

    // C layout (32-bit, 16x16): VGPR r -> M = r (lanes 0-15) or 8+r (16-31);
    // N = lane & 15.
    const int cn    = lane & 15;
    const int cmOff = (lane >> 4) * 8;
#pragma unroll
    for (int mi = 0; mi < 4; ++mi) {
#pragma unroll
        for (int ni = 0; ni < 4; ++ni) {
            float* cp = C + (size_t)(mBase + mi * 16 + cmOff) * OUT_F
                          + (nBase + ni * 16 + cn);
#pragma unroll
            for (int r = 0; r < 8; ++r)
                cp[(size_t)r * OUT_F] = acc[mi][ni][r];
        }
    }
}

// ---------------------------------------------------------------------------
// Host launcher
// ---------------------------------------------------------------------------
extern "C" void kernel_launch(void* const* d_in, const int* in_sizes, int n_in,
                              void* d_out, int out_size, void* d_ws, size_t ws_size,
                              hipStream_t stream) {
    const float* x  = (const float*)d_in[0];   // [4, 2048, IN_F] f32
    const int*   qw = (const int*)  d_in[1];   // [W_BIT, OUT_F*IN_F/8] int32 bytes
    const float* u  = (const float*)d_in[2];   // [W_BIT, OUT_F, RK] f32
    const float* vt = (const float*)d_in[3];   // [W_BIT, RK, IN_F] f32
    float* out = (float*)d_out;                // [4, 2048, OUT_F] f32

    _Float16* x16 = (_Float16*)d_ws;                        // 8192*4096*2  = 64 MiB
    _Float16* w16 = x16 + (size_t)M_TOT * IN_F;             // 11008*4096*2 = 86 MiB

    // 1) x -> f16   (33,554,432 elems / 1024 per block = 32768 blocks, exact)
    cvt_x_kernel<<<(M_TOT * (size_t)IN_F) / 1024, 256, 0, stream>>>(x, x16);

    // 2) reconstruct w in f16
    recon_w_kernel<<<OUT_F, 256, 0, stream>>>(qw, u, vt, w16);

    // 3) WMMA GEMM: grid = (11008/256, 8192/128) = (43, 64)
    dim3 grid(OUT_F / 256, M_TOT / 128);
    gemm_f16_kernel<<<grid, 256, 0, stream>>>(x16, w16, out);
}
